// HybridAttentionLayer_40596030882495
// MI455X (gfx1250) — compile-verified, matrix operand order checked by reading, not run on hardware
//
#include <hip/hip_runtime.h>

typedef _Float16 h16;
typedef __attribute__((ext_vector_type(16))) _Float16 v16h;
typedef __attribute__((ext_vector_type(8)))  _Float16 v8h;
typedef __attribute__((ext_vector_type(8)))  float    v8f;

#define B_  4
#define S_  2048
#define D_  1024
#define H_  16
#define HD_ 64
#define F_  64
#define BH_ (B_*H_)
#define M_  (B_*S_)

// ---------------------------------------------------------------------------
// WMMA helpers (CDNA5 16x16x32 f16 -> f32)
// A fragment (16x32, row-major source): lane l holds row (l&15); elements
// 0-7 = K {kbase..kbase+7}, 8-15 = K {16+kbase..}, kbase = (l>>4)*8.
// => two contiguous 16B loads per lane.  B operand must be given as Bt[N][K].
// D tile: lane l holds col n=(l&15); element r holds row m = r + (l>>4)*8.
// ---------------------------------------------------------------------------
__device__ __forceinline__ v16h load_frag(const h16* p, int ld) {
  const int l = threadIdx.x & 31;
  const h16* r = p + (l & 15) * ld + ((l >> 4) << 3);
  v8h lo = *(const v8h*)(r);
  v8h hi = *(const v8h*)(r + 16);
  v16h f;
#pragma unroll
  for (int i = 0; i < 8; ++i) { f[i] = lo[i]; f[i + 8] = hi[i]; }
  return f;
}

__device__ __forceinline__ v8f wmma32(v16h a, v16h b, v8f c) {
  return __builtin_amdgcn_wmma_f32_16x16x32_f16(false, a, false, b, (short)0, c,
                                                false, false);
}

__device__ __forceinline__ v8f vzero8() {
  v8f z = {0.f, 0.f, 0.f, 0.f, 0.f, 0.f, 0.f, 0.f};
  return z;
}

// ---------------------------------------------------------------------------
// CDNA5 async global->LDS copy (16B per lane), tracked with ASYNCcnt.
// LDS byte address = low 32 bits of the generic pointer (LDS aperture keeps
// the offset in addr[31:0]).  GV addressing: 64-bit vaddr, saddr = off.
// ---------------------------------------------------------------------------
__device__ __forceinline__ void async_copy_b128(h16* lds_dst, const h16* gsrc) {
  asm volatile("global_load_async_to_lds_b128 %0, %1, off"
               :
               : "v"((unsigned)(unsigned long long)lds_dst), "v"(gsrc)
               : "memory");
}

__device__ __forceinline__ void wait_async_zero() {
  asm volatile("s_wait_asynccnt 0x0" ::: "memory");
}

// ---------------------------------------------------------------------------
// Prep: f32 -> f16 convert, and transpose-convert (W[k][n] -> Wt[n][k]).
// ---------------------------------------------------------------------------
__global__ void cvt_f16_kernel(const float* __restrict__ src, h16* __restrict__ dst,
                               int n) {
  int i = blockIdx.x * 256 + threadIdx.x;
  if (i < n) dst[i] = (h16)src[i];
}

__global__ void transpose_f16_kernel(const float* __restrict__ src,
                                     h16* __restrict__ dst, int K, int N) {
  int i = blockIdx.x * 256 + threadIdx.x;
  if (i >= K * N) return;
  int k = i / N, n = i - k * N;
  dst[(size_t)n * K + k] = (h16)src[i];
}

// ---------------------------------------------------------------------------
// GEMM: C[M=8192][1024] = Xh[M][1024] @ Wt (stored [N][K]), f16 store.
// mode 0: out[b][h][s][d]; mode 1: out[b][h][d][s].
// Block: 128 thr = 4 waves (2x2), block tile 64x64, wave tile 32x32.
// Double-buffered LDS fed by global_load_async_to_lds_b128; one barrier/step.
// ---------------------------------------------------------------------------
__global__ void gemm1024_kernel(const h16* __restrict__ A, const h16* __restrict__ Bt,
                                h16* __restrict__ out, int mode) {
  __shared__ __align__(16) h16 As[2][64][48];
  __shared__ __align__(16) h16 Bs[2][64][48];
  const int tid = threadIdx.x, wave = tid >> 5, lane = tid & 31;
  const int wm = wave >> 1, wn = wave & 1;
  const int bm = blockIdx.x * 64, bn = blockIdx.y * 64;
  const int lr = tid >> 1, lc = (tid & 1) * 16;

  const h16* gA = A + (size_t)(bm + lr) * 1024 + lc;
  const h16* gB = Bt + (size_t)(bn + lr) * 1024 + lc;

  v8f acc[2][2];
  acc[0][0] = vzero8(); acc[0][1] = vzero8();
  acc[1][0] = vzero8(); acc[1][1] = vzero8();

  // prologue: stage K-step 0 into buffer 0
  async_copy_b128(&As[0][lr][lc], gA);
  async_copy_b128(&As[0][lr][lc + 8], gA + 8);
  async_copy_b128(&Bs[0][lr][lc], gB);
  async_copy_b128(&Bs[0][lr][lc + 8], gB + 8);
  wait_async_zero();
  __syncthreads();

  for (int i = 0; i < 32; ++i) {
    const int cur = i & 1;
    if (i + 1 < 32) {  // async-stage next K slice into the other buffer
      const int nxt = cur ^ 1;
      const h16* ga = gA + (i + 1) * 32;
      const h16* gb = gB + (i + 1) * 32;
      async_copy_b128(&As[nxt][lr][lc], ga);
      async_copy_b128(&As[nxt][lr][lc + 8], ga + 8);
      async_copy_b128(&Bs[nxt][lr][lc], gb);
      async_copy_b128(&Bs[nxt][lr][lc + 8], gb + 8);
      __builtin_prefetch(ga + 32, 0, 3);
      __builtin_prefetch(gb + 32, 0, 3);
    }
    v16h af0 = load_frag(&As[cur][wm * 32][0], 48);
    v16h af1 = load_frag(&As[cur][wm * 32 + 16][0], 48);
    v16h bf0 = load_frag(&Bs[cur][wn * 32][0], 48);
    v16h bf1 = load_frag(&Bs[cur][wn * 32 + 16][0], 48);
    acc[0][0] = wmma32(af0, bf0, acc[0][0]);
    acc[0][1] = wmma32(af0, bf1, acc[0][1]);
    acc[1][0] = wmma32(af1, bf0, acc[1][0]);
    acc[1][1] = wmma32(af1, bf1, acc[1][1]);
    if (i + 1 < 32) wait_async_zero();
    __syncthreads();
  }

  const int col = lane & 15, rbase = (lane >> 4) << 3;
#pragma unroll
  for (int im = 0; im < 2; ++im)
#pragma unroll
    for (int in = 0; in < 2; ++in)
#pragma unroll
      for (int r = 0; r < 8; ++r) {
        int gm = bm + wm * 32 + im * 16 + rbase + r;
        int gn = bn + wn * 32 + in * 16 + col;
        int b = gm >> 11, s = gm & (S_ - 1);
        int h = gn >> 6, d = gn & 63;
        size_t idx = (mode == 0)
                         ? (((size_t)(b * H_ + h) * S_ + s) * HD_ + d)
                         : (((size_t)(b * H_ + h) * HD_ + d) * S_ + s);
        out[idx] = (h16)acc[im][in][r];
      }
}

// ---------------------------------------------------------------------------
// Flash (softmax) attention. One wave per 16-query tile; 32-key chunks.
// q,k: [BH][S][64] f16 ; vT: [BH][64][S] f16 ; flash out: [B][S][1024] f16.
// P re-fragmentation uses a per-wave-private LDS tile; intra-wave LDS ops are
// architecturally in-order, so a compiler fence replaces hardware barriers.
// ---------------------------------------------------------------------------
__global__ void flash_attn_kernel(const h16* __restrict__ q, const h16* __restrict__ k,
                                  const h16* __restrict__ vT, h16* __restrict__ flash) {
  __shared__ __align__(16) h16 p_lds[4][16][32];
  const int wave = threadIdx.x >> 5, lane = threadIdx.x & 31;
  const int bh = blockIdx.y, b = bh >> 4, h = bh & 15;
  const int qt = blockIdx.x * 4 + wave;  // 16-row query tile
  const h16* qb = q + ((size_t)bh * S_ + qt * 16) * HD_;
  const h16* kb = k + (size_t)bh * S_ * HD_;
  const h16* vb = vT + (size_t)bh * HD_ * S_;

  const v16h qa0 = load_frag(qb, HD_);
  const v16h qa1 = load_frag(qb + 32, HD_);

  v8f acc[4];
  float rowmax[8], rowsum[8];
#pragma unroll
  for (int t = 0; t < 4; ++t) acc[t] = vzero8();
#pragma unroll
  for (int r = 0; r < 8; ++r) { rowmax[r] = -1e30f; rowsum[r] = 0.f; }

  const int col = lane & 15, rbase = (lane >> 4) << 3;
  const float scale = 0.125f;  // 1/sqrt(64)

  for (int c = 0; c < S_ / 32; ++c) {
    const h16* kc = kb + (size_t)c * 32 * HD_;
    v8f s0 = vzero8(), s1 = vzero8();
    s0 = wmma32(qa0, load_frag(kc, HD_), s0);
    s0 = wmma32(qa1, load_frag(kc + 32, HD_), s0);
    s1 = wmma32(qa0, load_frag(kc + 16 * HD_, HD_), s1);
    s1 = wmma32(qa1, load_frag(kc + 16 * HD_ + 32, HD_), s1);

    float corr[8];
#pragma unroll
    for (int r = 0; r < 8; ++r) {
      float a = s0[r] * scale, bb = s1[r] * scale;
      float m = fmaxf(a, bb);
      m = fmaxf(m, __shfl_xor(m, 1));
      m = fmaxf(m, __shfl_xor(m, 2));
      m = fmaxf(m, __shfl_xor(m, 4));
      m = fmaxf(m, __shfl_xor(m, 8));            // 16-lane group == score row
      float nm = fmaxf(rowmax[r], m);
      float p0 = __expf(a - nm), p1 = __expf(bb - nm);
      s0[r] = p0; s1[r] = p1;
      float ps = p0 + p1;
      ps += __shfl_xor(ps, 1);
      ps += __shfl_xor(ps, 2);
      ps += __shfl_xor(ps, 4);
      ps += __shfl_xor(ps, 8);
      corr[r] = __expf(rowmax[r] - nm);
      rowsum[r] = rowsum[r] * corr[r] + ps;
      rowmax[r] = nm;
    }
#pragma unroll
    for (int t = 0; t < 4; ++t)
#pragma unroll
      for (int r = 0; r < 8; ++r) acc[t][r] *= corr[r];

    // re-fragment P (16x32) through the per-wave LDS tile (in-order per wave)
#pragma unroll
    for (int r = 0; r < 8; ++r) {
      p_lds[wave][rbase + r][col] = (h16)s0[r];
      p_lds[wave][rbase + r][16 + col] = (h16)s1[r];
    }
    __builtin_amdgcn_wave_barrier();
    v16h pf = load_frag(&p_lds[wave][0][0], 32);
    __builtin_amdgcn_wave_barrier();

    const h16* vc = vb + c * 32;
#pragma unroll
    for (int t = 0; t < 4; ++t)
      acc[t] = wmma32(pf, load_frag(vc + (size_t)t * 16 * S_, S_), acc[t]);
  }

#pragma unroll
  for (int r = 0; r < 8; ++r) {
    int srow = qt * 16 + rbase + r;
    float inv = 1.0f / rowsum[r];
    h16* o = flash + ((size_t)b * S_ + srow) * D_ + h * HD_ + col;
    o[0]  = (h16)(acc[0][r] * inv);
    o[16] = (h16)(acc[1][r] * inv);
    o[32] = (h16)(acc[2][r] * inv);
    o[48] = (h16)(acc[3][r] * inv);
  }
}

// ---------------------------------------------------------------------------
// Feature map: out = relu(inp @ fm_w + fm_b) per head.  inp [BH][S][64],
// fmWt [64(out)][64(in)].  mode 0: qf [BH][S][64]; mode 1: kfT [BH][64][S].
// ---------------------------------------------------------------------------
__global__ void featmap_kernel(const h16* __restrict__ inp, const h16* __restrict__ fmWt,
                               const float* __restrict__ bias, h16* __restrict__ out,
                               int mode) {
  const int wave = threadIdx.x >> 5, lane = threadIdx.x & 31;
  const int bh = blockIdx.y;
  const int mt = blockIdx.x * 4 + wave;
  const h16* ab = inp + ((size_t)bh * S_ + mt * 16) * HD_;
  v16h a0 = load_frag(ab, HD_), a1 = load_frag(ab + 32, HD_);
  v8f acc[4];
#pragma unroll
  for (int t = 0; t < 4; ++t) {
    acc[t] = vzero8();
    acc[t] = wmma32(a0, load_frag(fmWt + t * 16 * 64, 64), acc[t]);
    acc[t] = wmma32(a1, load_frag(fmWt + t * 16 * 64 + 32, 64), acc[t]);
  }
  const int col = lane & 15, rbase = (lane >> 4) << 3;
#pragma unroll
  for (int t = 0; t < 4; ++t)
#pragma unroll
    for (int r = 0; r < 8; ++r) {
      int f = t * 16 + col, s = mt * 16 + rbase + r;
      float v = fmaxf(acc[t][r] + bias[f], 0.0f);
      size_t idx = (mode == 0) ? (((size_t)bh * S_ + s) * F_ + f)
                               : (((size_t)bh * F_ + f) * S_ + s);
      out[idx] = (h16)v;
    }
}

// kv[f][d] = sum_s kf[s][f]*v[s][d]; A=kfT [BH][64][S], Bt=vlT [BH][64][S];
// store transposed: kvT [BH][d][f].  Block = 4 waves, one block per (b,h).
__global__ void kv_kernel(const h16* __restrict__ kfT, const h16* __restrict__ vlT,
                          h16* __restrict__ kvT) {
  const int wave = threadIdx.x >> 5, lane = threadIdx.x & 31;
  const int bh = blockIdx.x, f0 = wave * 16;
  const h16* ab = kfT + ((size_t)bh * F_ + f0) * S_;
  const h16* bb = vlT + (size_t)bh * HD_ * S_;
  v8f acc[4];
#pragma unroll
  for (int t = 0; t < 4; ++t) acc[t] = vzero8();
  for (int ks = 0; ks < S_; ks += 32) {
    v16h a = load_frag(ab + ks, S_);
#pragma unroll
    for (int t = 0; t < 4; ++t)
      acc[t] = wmma32(a, load_frag(bb + (size_t)t * 16 * S_ + ks, S_), acc[t]);
  }
  const int col = lane & 15, rbase = (lane >> 4) << 3;
#pragma unroll
  for (int t = 0; t < 4; ++t)
#pragma unroll
    for (int r = 0; r < 8; ++r) {
      int d = t * 16 + col, f = f0 + rbase + r;
      kvT[((size_t)bh * HD_ + d) * F_ + f] = (h16)acc[t][r];
    }
}

__global__ void ksum_kernel(const h16* __restrict__ kfT, float* __restrict__ ksum) {
  int id = blockIdx.x * 256 + threadIdx.x;
  if (id >= BH_ * F_) return;
  int bh = id >> 6, f = id & 63;
  const v8h* row = (const v8h*)(kfT + ((size_t)bh * F_ + f) * S_);
  float s = 0.f;
  for (int i = 0; i < S_ / 8; ++i) {
    v8h v = row[i];
#pragma unroll
    for (int j = 0; j < 8; ++j) s += (float)v[j];
  }
  ksum[id] = s;
}

__global__ void norm_kernel(const h16* __restrict__ qf, const float* __restrict__ ksum,
                            float* __restrict__ norm) {
  int id = blockIdx.x * 256 + threadIdx.x;
  if (id >= B_ * S_ * H_) return;
  int b = id >> 15, rem = id & 32767, s = rem >> 4, h = rem & 15;
  const h16* qr = qf + (((size_t)(b * H_ + h) * S_) + s) * F_;
  const float* kr = ksum + (b * H_ + h) * F_;
  float acc = 0.f;
  for (int f = 0; f < F_; ++f) acc += (float)qr[f] * kr[f];
  norm[id] = acc;
}

// lin = (qf @ kv) / max(norm,1e-6) -> linN [B][S][1024] f16.
__global__ void lin_kernel(const h16* __restrict__ qf, const h16* __restrict__ kvT,
                           const float* __restrict__ norm, h16* __restrict__ linN) {
  const int wave = threadIdx.x >> 5, lane = threadIdx.x & 31;
  const int bh = blockIdx.y, b = bh >> 4, h = bh & 15;
  const int mt = blockIdx.x * 4 + wave;
  const h16* ab = qf + ((size_t)bh * S_ + mt * 16) * F_;
  const h16* kv = kvT + (size_t)bh * HD_ * F_;
  v16h a0 = load_frag(ab, F_), a1 = load_frag(ab + 32, F_);
  v8f acc[4];
#pragma unroll
  for (int t = 0; t < 4; ++t) {
    acc[t] = vzero8();
    acc[t] = wmma32(a0, load_frag(kv + t * 16 * F_, F_), acc[t]);
    acc[t] = wmma32(a1, load_frag(kv + t * 16 * F_ + 32, F_), acc[t]);
  }
  const int col = lane & 15, rbase = (lane >> 4) << 3;
#pragma unroll
  for (int t = 0; t < 4; ++t)
#pragma unroll
    for (int r = 0; r < 8; ++r) {
      int d = t * 16 + col, s = mt * 16 + rbase + r;
      float nr = fmaxf(norm[((size_t)b * S_ + s) * H_ + h], 1e-6f);
      linN[((size_t)b * S_ + s) * D_ + h * HD_ + d] = (h16)(acc[t][r] / nr);
    }
}

__global__ void gate_kernel(const float* __restrict__ x, const float* __restrict__ gw,
                            const float* __restrict__ gb, float* __restrict__ gate) {
  int m = blockIdx.x * 256 + threadIdx.x;
  if (m >= M_) return;
  const float4* xr = (const float4*)(x + (size_t)m * D_);
  const float4* wr = (const float4*)gw;
  float s = 0.f;
  for (int i = 0; i < D_ / 4; ++i) {
    float4 a = xr[i], w = wr[i];
    s += a.x * w.x + a.y * w.y + a.z * w.z + a.w * w.w;
  }
  s += gb[0];
  gate[m] = 1.0f / (1.0f + __expf(-s));
}

// ---------------------------------------------------------------------------
// Fused output: out = g*(flash@fo_w + fo_b) + (1-g)*(linN@lo_w + lo_b).
// ---------------------------------------------------------------------------
__global__ void final_blend_kernel(const h16* __restrict__ Af, const h16* __restrict__ Al,
                                   const h16* __restrict__ BtF, const h16* __restrict__ BtL,
                                   const float* __restrict__ fob, const float* __restrict__ lob,
                                   const float* __restrict__ gate, float* __restrict__ out) {
  __shared__ __align__(16) h16 AsF[64][48], AsL[64][48], BsF[64][48], BsL[64][48];
  const int tid = threadIdx.x, wave = tid >> 5, lane = tid & 31;
  const int wm = wave >> 1, wn = wave & 1;
  const int bm = blockIdx.x * 64, bn = blockIdx.y * 64;
  const int lr = tid >> 1, lc = (tid & 1) * 16;

  v8f aF[2][2], aL[2][2];
#pragma unroll
  for (int i = 0; i < 2; ++i)
#pragma unroll
    for (int j = 0; j < 2; ++j) { aF[i][j] = vzero8(); aL[i][j] = vzero8(); }

  for (int k0 = 0; k0 < 1024; k0 += 32) {
    const h16* g0 = Af + (size_t)(bm + lr) * 1024 + k0 + lc;
    const h16* g1 = Al + (size_t)(bm + lr) * 1024 + k0 + lc;
    const h16* g2 = BtF + (size_t)(bn + lr) * 1024 + k0 + lc;
    const h16* g3 = BtL + (size_t)(bn + lr) * 1024 + k0 + lc;
    v8h r0a = *(const v8h*)g0, r0b = *(const v8h*)(g0 + 8);
    v8h r1a = *(const v8h*)g1, r1b = *(const v8h*)(g1 + 8);
    v8h r2a = *(const v8h*)g2, r2b = *(const v8h*)(g2 + 8);
    v8h r3a = *(const v8h*)g3, r3b = *(const v8h*)(g3 + 8);
    __builtin_prefetch(g0 + 32, 0, 3);
    __builtin_prefetch(g2 + 32, 0, 3);
    __syncthreads();
    *(v8h*)&AsF[lr][lc] = r0a; *(v8h*)&AsF[lr][lc + 8] = r0b;
    *(v8h*)&AsL[lr][lc] = r1a; *(v8h*)&AsL[lr][lc + 8] = r1b;
    *(v8h*)&BsF[lr][lc] = r2a; *(v8h*)&BsF[lr][lc + 8] = r2b;
    *(v8h*)&BsL[lr][lc] = r3a; *(v8h*)&BsL[lr][lc + 8] = r3b;
    __syncthreads();
    v16h af0 = load_frag(&AsF[wm * 32][0], 48), af1 = load_frag(&AsF[wm * 32 + 16][0], 48);
    v16h al0 = load_frag(&AsL[wm * 32][0], 48), al1 = load_frag(&AsL[wm * 32 + 16][0], 48);
    v16h bf0 = load_frag(&BsF[wn * 32][0], 48), bf1 = load_frag(&BsF[wn * 32 + 16][0], 48);
    v16h bl0 = load_frag(&BsL[wn * 32][0], 48), bl1 = load_frag(&BsL[wn * 32 + 16][0], 48);
    aF[0][0] = wmma32(af0, bf0, aF[0][0]);
    aF[0][1] = wmma32(af0, bf1, aF[0][1]);
    aF[1][0] = wmma32(af1, bf0, aF[1][0]);
    aF[1][1] = wmma32(af1, bf1, aF[1][1]);
    aL[0][0] = wmma32(al0, bl0, aL[0][0]);
    aL[0][1] = wmma32(al0, bl1, aL[0][1]);
    aL[1][0] = wmma32(al1, bl0, aL[1][0]);
    aL[1][1] = wmma32(al1, bl1, aL[1][1]);
  }

  const int col = lane & 15, rbase = (lane >> 4) << 3;
#pragma unroll
  for (int im = 0; im < 2; ++im)
#pragma unroll
    for (int in = 0; in < 2; ++in)
#pragma unroll
      for (int r = 0; r < 8; ++r) {
        int gm = bm + wm * 32 + im * 16 + rbase + r;
        int gn = bn + wn * 32 + in * 16 + col;
        float g = gate[gm];
        float o = g * (aF[im][in][r] + fob[gn]) + (1.f - g) * (aL[im][in][r] + lob[gn]);
        out[(size_t)gm * D_ + gn] = o;
      }
}

// ---------------------------------------------------------------------------
extern "C" void kernel_launch(void* const* d_in, const int* in_sizes, int n_in,
                              void* d_out, int out_size, void* d_ws, size_t ws_size,
                              hipStream_t stream) {
  const float* x    = (const float*)d_in[0];
  const float* fq   = (const float*)d_in[1];
  const float* fk   = (const float*)d_in[2];
  const float* fv   = (const float*)d_in[3];
  const float* fo_w = (const float*)d_in[4];
  const float* fo_b = (const float*)d_in[5];
  const float* lq   = (const float*)d_in[6];
  const float* lk   = (const float*)d_in[7];
  const float* lv   = (const float*)d_in[8];
  const float* lo_w = (const float*)d_in[9];
  const float* lo_b = (const float*)d_in[10];
  const float* fm_w = (const float*)d_in[11];
  const float* fm_b = (const float*)d_in[12];
  const float* gw   = (const float*)d_in[13];
  const float* gb   = (const float*)d_in[14];
  float* out = (float*)d_out;

  char* p = (char*)d_ws;
  auto take = [&](size_t n) { char* r = p; p += (n + 255) & ~(size_t)255; return r; };
  const size_t wsz = (size_t)D_ * D_ * sizeof(h16);     // 2 MB per weight
  const size_t big = (size_t)M_ * D_ * sizeof(h16);     // 16 MB

  h16* WtFq = (h16*)take(wsz); h16* WtFk = (h16*)take(wsz);
  h16* WtFv = (h16*)take(wsz); h16* WtLq = (h16*)take(wsz);
  h16* WtLk = (h16*)take(wsz); h16* WtLv = (h16*)take(wsz);
  h16* WtFo = (h16*)take(wsz); h16* WtLo = (h16*)take(wsz);
  h16* fmWt = (h16*)take((size_t)HD_ * F_ * sizeof(h16));
  h16* xh   = (h16*)take(big);          // reused as kfT later
  h16* bufQ = (h16*)take(big);          // q   -> vlT
  h16* bufK = (h16*)take(big);          // k   -> lq/lk projection scratch
  h16* bufV = (h16*)take(big);          // vT  -> qf
  h16* flash = (h16*)take(big);
  h16* linN  = (h16*)take(big);
  h16* kvT  = (h16*)take((size_t)BH_ * HD_ * F_ * sizeof(h16));
  float* ksum  = (float*)take((size_t)BH_ * F_ * sizeof(float));
  float* normb = (float*)take((size_t)B_ * S_ * H_ * sizeof(float));
  float* gateb = (float*)take((size_t)M_ * sizeof(float));

  // --- prep: convert x, transpose+convert all weights ---
  cvt_f16_kernel<<<(M_ * D_ + 255) / 256, 256, 0, stream>>>(x, xh, M_ * D_);
  const float* wsrc[8] = {fq, fk, fv, lq, lk, lv, fo_w, lo_w};
  h16* wdst[8] = {WtFq, WtFk, WtFv, WtLq, WtLk, WtLv, WtFo, WtLo};
  for (int i = 0; i < 8; ++i)
    transpose_f16_kernel<<<(D_ * D_ + 255) / 256, 256, 0, stream>>>(wsrc[i], wdst[i], D_, D_);
  transpose_f16_kernel<<<(HD_ * F_ + 255) / 256, 256, 0, stream>>>(fm_w, fmWt, HD_, F_);

  dim3 gGemm(M_ / 64, D_ / 64);
  dim3 gHead(S_ / 64, BH_);

  // --- flash branch ---
  gemm1024_kernel<<<gGemm, 128, 0, stream>>>(xh, WtFq, bufQ, 0);
  gemm1024_kernel<<<gGemm, 128, 0, stream>>>(xh, WtFk, bufK, 0);
  gemm1024_kernel<<<gGemm, 128, 0, stream>>>(xh, WtFv, bufV, 1);
  flash_attn_kernel<<<gHead, 128, 0, stream>>>(bufQ, bufK, bufV, flash);

  // --- linear branch (reuses q/k/v buffers after flash) ---
  gemm1024_kernel<<<gGemm, 128, 0, stream>>>(xh, WtLv, bufQ, 1);            // vlT
  gemm1024_kernel<<<gGemm, 128, 0, stream>>>(xh, WtLq, bufK, 0);            // ql
  featmap_kernel<<<gHead, 128, 0, stream>>>(bufK, fmWt, fm_b, bufV, 0);     // qf
  gemm1024_kernel<<<gGemm, 128, 0, stream>>>(xh, WtLk, bufK, 0);            // kl (last use of xh)
  featmap_kernel<<<gHead, 128, 0, stream>>>(bufK, fmWt, fm_b, xh, 1);       // kfT -> xh
  kv_kernel<<<BH_, 128, 0, stream>>>(xh, bufQ, kvT);
  ksum_kernel<<<(BH_ * F_ + 255) / 256, 256, 0, stream>>>(xh, ksum);
  norm_kernel<<<(B_ * S_ * H_ + 255) / 256, 256, 0, stream>>>(bufV, ksum, normb);
  lin_kernel<<<gHead, 128, 0, stream>>>(bufV, kvT, normb, linN);

  // --- gate + fused output projections + blend ---
  gate_kernel<<<(M_ + 255) / 256, 256, 0, stream>>>(x, gw, gb, gateb);
  final_blend_kernel<<<gGemm, 128, 0, stream>>>(flash, linN, WtFo, WtLo,
                                                fo_b, lo_b, gateb, out);
}
